// SuperHighPerformanceExtractor_45277545234759
// MI455X (gfx1250) — matrix operand order repaired and checked by reading, not compiled, not run on hardware
//
#include <hip/hip_runtime.h>
#include <hip/hip_bf16.h>

// Shapes from the reference
#define BB 32
#define TT 512
#define DD 512
#define FF 512

typedef __attribute__((ext_vector_type(16))) __bf16 v16bf;
typedef __attribute__((ext_vector_type(8)))  float  v8f;

// bf16 hi part of an fp32 (round-to-nearest-even)
static __device__ __forceinline__ __bf16 bf_hi(float x) { return (__bf16)x; }
// bf16 lo part: residual after removing hi, so hi+lo ~= x with ~2^-16 rel error
static __device__ __forceinline__ __bf16 bf_lo(float x, __bf16 hi) {
    return (__bf16)(x - (float)hi);
}

template <bool NT>
static __device__ __forceinline__ float ldw(const float* p) {
    if (NT) return __builtin_nontemporal_load(p);   // TH=NT: Wext streams once, don't pollute caches
    return *p;
}

static __device__ __forceinline__ void load_a_frag(const float* arow, int k0, int half,
                                                   v16bf& ahi, v16bf& alo) {
    // A fragment (16x32 bf16, ISA layout): lane holds row m;
    //   element j<8  -> K = k0 + 8*half + j
    //   element j>=8 -> K = k0 + 16 + 8*half + (j-8)
    const float4* p0 = reinterpret_cast<const float4*>(arow + k0 + 8 * half);
    const float4* p1 = reinterpret_cast<const float4*>(arow + k0 + 16 + 8 * half);
    const float4 r0 = p0[0], r1 = p0[1];
    const float4 r2 = p1[0], r3 = p1[1];
    const float av[16] = { r0.x, r0.y, r0.z, r0.w,  r1.x, r1.y, r1.z, r1.w,
                           r2.x, r2.y, r2.z, r2.w,  r3.x, r3.y, r3.z, r3.w };
    #pragma unroll
    for (int j = 0; j < 16; ++j) {
        const __bf16 h = bf_hi(av[j]);
        ahi[j] = h;
        alo[j] = bf_lo(av[j], h);
    }
}

// Unified 32xK @ KxF GEMM tile kernel (M fixed = 32 rows per blockIdx.y slice).
//   A row m:  A0 + blockIdx.y*a_ystride + m*a_mstride   (K contiguous fp32)
//   W     :  W0 + blockIdx.y*w_ystride                  (row-major K x F fp32)
//   Out   :  O0 + blockIdx.y*o_ystride + m*o_mstride + n
// Block: 256 threads = 8 waves covering a 32x128 output tile. Each wave owns one
// 16-col N tile and BOTH 16-row M tiles (two accumulators), so every B (Wext)
// element is loaded by exactly one wave.
// D = Ahi*Bhi + Ahi*Blo + Alo*Bhi  via V_WMMA_F32_16X16X32_BF16 (f32 accumulate):
// near-fp32 accuracy (error ~2^-16, the dropped lo*lo term) at bf16 matrix rate,
// which keeps both GEMMs on the 23.3 TB/s memory roof (~16-28 FLOP/byte here).
template <bool NT_W>
__global__ __launch_bounds__(256)
void gemm32_bf16split(const float* __restrict__ A0, size_t a_mstride, size_t a_ystride,
                      const float* __restrict__ W0, size_t w_ystride,
                      float* __restrict__ O0, size_t o_mstride, size_t o_ystride,
                      int K, int F)
{
    const int lane = threadIdx.x & 31;
    const int ni   = threadIdx.x >> 5;   // 8 N tiles (128 cols per block)
    const int half = lane >> 4;          // lane group 0: lanes 0-15, group 1: lanes 16-31
    const int l16  = lane & 15;
    const int n    = blockIdx.x * 128 + ni * 16 + l16;

    const float* abase = A0 + (size_t)blockIdx.y * a_ystride;
    const float* arow0 = abase + (size_t)(l16)      * a_mstride;  // M tile 0, row l16
    const float* arow1 = abase + (size_t)(l16 + 16) * a_mstride;  // M tile 1, row l16+16
    const float* W     = W0 + (size_t)blockIdx.y * w_ystride;

    v8f acc0 = {};
    v8f acc1 = {};

    for (int k0 = 0; k0 < K; k0 += 32) {
        v16bf a0hi{}, a0lo{}, a1hi{}, a1lo{}, bhi{}, blo{};

        load_a_frag(arow0, k0, half, a0hi, a0lo);
        load_a_frag(arow1, k0, half, a1hi, a1lo);

        // B fragment (32x16 bf16, ISA layout): lane holds column n;
        //   element j<8 -> K = k0 + 8*half + j; element j>=8 -> K = k0 + 16 + 8*half + (j-8)
        {
            const float* bp = W + (size_t)k0 * F + n;
            #pragma unroll
            for (int j = 0; j < 16; ++j) {
                const float x = ldw<NT_W>(bp + (size_t)(8 * half + j + (j >= 8 ? 8 : 0)) * F);
                const __bf16 h = bf_hi(x);
                bhi[j] = h;
                blo[j] = bf_lo(x, h);
            }
        }

        acc0 = __builtin_amdgcn_wmma_f32_16x16x32_bf16(false, a0hi, false, bhi, (short)0, acc0, false, false);
        acc0 = __builtin_amdgcn_wmma_f32_16x16x32_bf16(false, a0hi, false, blo, (short)0, acc0, false, false);
        acc0 = __builtin_amdgcn_wmma_f32_16x16x32_bf16(false, a0lo, false, bhi, (short)0, acc0, false, false);
        acc1 = __builtin_amdgcn_wmma_f32_16x16x32_bf16(false, a1hi, false, bhi, (short)0, acc1, false, false);
        acc1 = __builtin_amdgcn_wmma_f32_16x16x32_bf16(false, a1hi, false, blo, (short)0, acc1, false, false);
        acc1 = __builtin_amdgcn_wmma_f32_16x16x32_bf16(false, a1lo, false, bhi, (short)0, acc1, false, false);
    }

    // ---- D fragment (16x16 f32, ISA layout): VGPR v -> row (v + 8*half), col l16.
    // 32-bit offsets from a single base so the backend can use saddr+voffset addressing.
    float* obase = O0 + (size_t)blockIdx.y * o_ystride;
    const unsigned oms = (unsigned)o_mstride;
    #pragma unroll
    for (int v = 0; v < 8; ++v) {
        const unsigned mo = (unsigned)(v + 8 * half);
        obase[mo * oms + (unsigned)n]        = acc0[v];
        obase[(mo + 16) * oms + (unsigned)n] = acc1[v];
    }
}

// Per-(b,f) inclusive scan over t of xi, scale by 1/sqrt(t+1), gate with d_out in place.
// Unrolled x4 with all loads issued before the dependent adds to maximize MLP
// (the cumsum is a loop-carried dependence; the loads are not).
__global__ __launch_bounds__(128)
void scan_scale_gate(const float* __restrict__ xi, float* __restrict__ out, int T, int F)
{
    const int f = blockIdx.x * blockDim.x + threadIdx.x;
    const size_t colbase = (size_t)blockIdx.y * (size_t)T * F + f;
    float s = 0.0f;
    for (int t = 0; t < T; t += 4) {
        const size_t i0 = colbase + (size_t)t * F;
        const size_t i1 = i0 + F;
        const size_t i2 = i1 + F;
        const size_t i3 = i2 + F;
        const float x0 = __builtin_nontemporal_load(xi + i0);
        const float x1 = __builtin_nontemporal_load(xi + i1);
        const float x2 = __builtin_nontemporal_load(xi + i2);
        const float x3 = __builtin_nontemporal_load(xi + i3);
        const float g0 = out[i0];
        const float g1 = out[i1];
        const float g2 = out[i2];
        const float g3 = out[i3];
        s += x0; out[i0] = g0 * s * (1.0f / sqrtf((float)(t + 1)));
        s += x1; out[i1] = g1 * s * (1.0f / sqrtf((float)(t + 2)));
        s += x2; out[i2] = g2 * s * (1.0f / sqrtf((float)(t + 3)));
        s += x3; out[i3] = g3 * s * (1.0f / sqrtf((float)(t + 4)));
    }
}

extern "C" void kernel_launch(void* const* d_in, const int* in_sizes, int n_in,
                              void* d_out, int out_size, void* d_ws, size_t ws_size,
                              hipStream_t stream) {
    const float* inputs = (const float*)d_in[0];   // (B,T,D)
    const float* Wext   = (const float*)d_in[1];   // (T,D,F)
    const float* Wadj   = (const float*)d_in[2];   // (D,F)
    // d_in[3] = training flag, unused
    float* out = (float*)d_out;                    // (B,T,F)
    float* xi  = (float*)d_ws;                     // (B,T,F) scratch: 32 MB

    // Phase 1: xi[b,t,:] = inputs[b,t,:] @ Wext[t]   (one block per (t, 128-col group))
    //   A row m (= batch b): inputs + t*D + m*(T*D)
    //   W: Wext + t*(D*F); Out row m: xi + t*F + m*(T*F)
    dim3 blk(256);
    dim3 g1(FF / 128, TT);
    gemm32_bf16split<true><<<g1, blk, 0, stream>>>(
        inputs, (size_t)TT * DD, (size_t)DD,
        Wext,   (size_t)DD * FF,
        xi,     (size_t)TT * FF, (size_t)FF,
        DD, FF);

    // Phase 2: gate = (inputs reshaped (B*T, D)) @ Wadj  -> d_out
    //   A row m within 32-row slice y: inputs + y*32*D + m*D
    dim3 g2(FF / 128, (BB * TT) / 32);
    gemm32_bf16split<false><<<g2, blk, 0, stream>>>(
        inputs, (size_t)DD, (size_t)32 * DD,
        Wadj,   (size_t)0,
        out,    (size_t)FF, (size_t)32 * FF,
        DD, FF);

    // Phase 3: inclusive cumsum over t + 1/sqrt(t+1) scaling + gating (in place on d_out)
    dim3 g3(FF / 128, BB);
    scan_scale_gate<<<g3, dim3(128), 0, stream>>>(xi, out, TT, FF);
}
